// SpatialDecoder_29248727286193
// MI455X (gfx1250) — matrix operand order repaired and checked by reading, not compiled, not run on hardware
//
#include <hip/hip_runtime.h>
#include <hip/hip_bf16.h>

// ---------------------------------------------------------------------------
// SpatialDecoder on MI455X (gfx1250, wave32, WMMA + async LDS staging)
//
// Dominant op refactored: fold W_mlp in front -> Y[b,s,o,v] (M=64), then
//   out[b,o,w] = b_mlp[o] + sum_s sum_v Y[b,s,o,v] * adj[s,w,v]
// Big GEMM: bf16 WMMA (v_wmma_f32_16x16x32_bf16) with f32 accum; A/B tiles
// double-buffered in LDS via global_load_async_to_lds_b128 (ASYNCcnt).
// Block tile 64x256, wave tile 32x64 (8 WMMA per 12 ds_load_b128).
//
// Workspace layout (needs >= 128 MiB):
//   [0)                adj_bf16 : S*N*N      ushort  (67,108,864 B)
//   [+67108864)        Y_bf16   : B*S*64*N   ushort  (33,554,432 B)
//   [+100663296)       out_f32  : B*64*N     float   (33,554,432 B)
// ---------------------------------------------------------------------------

#define BB     32
#define NN     4096
#define SS     2
#define DM     64
#define DIN    65      // C_X + D_MODEL

#define BN     256             // block tile columns
#define KSTG   64              // K-halves per LDS stage (2 WMMA sub-steps)
#define LDA    72              // padded row stride (halves) for A tile in LDS
#define LDB    72              // padded col stride (halves) for B tile in LDS

typedef __attribute__((ext_vector_type(16))) __bf16 bf16x16;
typedef __attribute__((ext_vector_type(8)))  float  f32x8;
typedef __attribute__((ext_vector_type(4)))  int    i32x4;

union FragAB { bf16x16 v; uint4 q[2]; };
union FragC  { f32x8   v; float f[8]; };

#define AS1 __attribute__((address_space(1)))
#define AS3 __attribute__((address_space(3)))

__device__ __forceinline__ unsigned short f2bf(float f) {
    unsigned int u = __float_as_uint(f);
    u += 0x7fffu + ((u >> 16) & 1u);        // round-to-nearest-even
    return (unsigned short)(u >> 16);
}

// 16B per-lane async copy: global -> LDS, tracked by ASYNCcnt.
__device__ __forceinline__ void async_cp16(const unsigned short* g, unsigned short* l) {
#if __has_builtin(__builtin_amdgcn_global_load_async_to_lds_b128)
    __builtin_amdgcn_global_load_async_to_lds_b128(
        (AS1 i32x4*)(g), (AS3 i32x4*)(l), 0, 0);
#else
    unsigned      loff = (unsigned)(unsigned long long)(AS3 char*)l;
    unsigned long long ga = (unsigned long long)g;
    asm volatile("global_load_async_to_lds_b128 %0, %1, off"
                 :: "v"(loff), "v"(ga) : "memory");
#endif
}

#if __has_builtin(__builtin_amdgcn_s_wait_asynccnt)
#define WAIT_ASYNC(n) __builtin_amdgcn_s_wait_asynccnt(n)
#else
#define WAIT_ASYNC(n) asm volatile("s_wait_asynccnt %0" :: "i"(n))
#endif

#define WMMA_BF16(A, Bm, C) \
    __builtin_amdgcn_wmma_f32_16x16x32_bf16(false, (A), false, (Bm), (short)0, (C), false, false)

// ---------------------------------------------------------------------------
// Kernel 1: adj f32 -> bf16, layout preserved [s][w][v] (v contiguous)
// ---------------------------------------------------------------------------
__global__ __launch_bounds__(256) void k_cvt_adj(const float* __restrict__ adj,
                                                 unsigned short* __restrict__ adjb) {
    size_t i = ((size_t)blockIdx.x * 256 + threadIdx.x) * 8;
    float4 a = *(const float4*)(adj + i);
    float4 c = *(const float4*)(adj + i + 4);
    uint4 o;
    o.x = (unsigned)f2bf(a.x) | ((unsigned)f2bf(a.y) << 16);
    o.y = (unsigned)f2bf(a.z) | ((unsigned)f2bf(a.w) << 16);
    o.z = (unsigned)f2bf(c.x) | ((unsigned)f2bf(c.y) << 16);
    o.w = (unsigned)f2bf(c.z) | ((unsigned)f2bf(c.w) << 16);
    *(uint4*)(adjb + i) = o;
}

// ---------------------------------------------------------------------------
// Kernel 2: Y[b, r=s*64+o, v] = sum_c W_mlp[o, s*65+c] * x_in[b,c,v]  (bf16 out)
// ---------------------------------------------------------------------------
__global__ __launch_bounds__(256) void k_prep_y(const float* __restrict__ x,
                                                const float* __restrict__ h,
                                                const float* __restrict__ Wmlp,
                                                unsigned short* __restrict__ Yb) {
    __shared__ float xin[DIN * 128];
    const int b   = blockIdx.y;
    const int v0  = blockIdx.x * 128;
    const int tid = threadIdx.x;

    for (int idx = tid; idx < DIN * 128; idx += 256) {
        int k = idx >> 7, c = idx & 127;
        xin[idx] = (k == 0) ? x[(size_t)b * NN + v0 + c]
                            : h[((size_t)b * DM + (k - 1)) * NN + v0 + c];
    }
    __syncthreads();

    const int c  = tid & 127;
    const int rh = tid >> 7;
    for (int rr = 0; rr < 64; ++rr) {
        int r = rh * 64 + rr;              // r = s*64 + o
        int s = r >> 6, o = r & 63;
        const float* wrow = Wmlp + o * (SS * DIN) + s * DIN;
        float acc = 0.f;
        #pragma unroll
        for (int k = 0; k < DIN; ++k) acc += wrow[k] * xin[k * 128 + c];
        Yb[((size_t)b * (SS * DM) + r) * NN + v0 + c] = f2bf(acc);
    }
}

// ---------------------------------------------------------------------------
// Kernel 3: main GEMM with double-buffered async LDS staging.
// Block 256 thr = 8 waves; tile 64 rows x 256 cols.
// Wave (mi = wv&1, nj = wv>>1): rows [mi*32,+32), cols [nj*64,+64) -> 2x4 tiles.
// Stage = 64 K-halves: A tile 64x64 (9216 B), B tile 256x64 (36864 B) padded.
// ---------------------------------------------------------------------------
__global__ __launch_bounds__(256) void k_gemm(const unsigned short* __restrict__ Yb,
                                              const unsigned short* __restrict__ adjb,
                                              float* __restrict__ outw) {
    __shared__ __align__(16) unsigned short sA[2][DM * LDA];     // [row][k]
    __shared__ __align__(16) unsigned short sB[2][BN * LDB];     // [col][k]

    const int tid  = threadIdx.x;
    const int b    = blockIdx.y;
    const int lane = tid & 31;
    const int wv   = tid >> 5;
    const int mi   = wv & 1;
    const int nj   = wv >> 1;
    const int wbase0 = blockIdx.x * BN;
    const int l15  = lane & 15;
    const int lhi  = lane >> 4;

    // ---- per-thread async-copy mapping (stage-invariant parts) -------------
    // A: 64 rows x 8 chunks(16B) = 512 transfers -> 2 per thread.
    const int ar0 = (tid + 0)   >> 3, ac0 = (tid + 0)   & 7;
    const int ar1 = (tid + 256) >> 3, ac1 = (tid + 256) & 7;
    const unsigned short* gA0 = Yb + ((size_t)(b * SS) * DM + ar0) * NN + ac0 * 8;
    const unsigned short* gA1 = Yb + ((size_t)(b * SS) * DM + ar1) * NN + ac1 * 8;
    const int lA0 = ar0 * LDA + ac0 * 8, lA1 = ar1 * LDA + ac1 * 8;
    // B: 256 cols x 8 chunks = 2048 transfers -> 8 per thread.
    const int brr = tid >> 3, bcc = tid & 7;      // base (j=0); j adds 32 cols
    const unsigned short* gB0 = adjb + ((size_t)(wbase0 + brr)) * NN + bcc * 8;
    const int lB0 = brr * LDB + bcc * 8;          // j adds 32*LDB halves

    const size_t sStrA = (size_t)DM * NN;         // +s -> next Y[b][s] block
    const size_t sStrB = (size_t)NN * NN;         // +s -> next adj support

    // ---- accumulators: 2 row-blocks x 4 col-tiles --------------------------
    FragC acc[2][4];
    #pragma unroll
    for (int r = 0; r < 2; ++r)
        #pragma unroll
        for (int t = 0; t < 4; ++t)
            acc[r][t].v = (f32x8){0,0,0,0,0,0,0,0};

    const int NSTAGE = SS * (NN / KSTG);   // 128

    // issue stage 0
    {
        async_cp16(gA0, &sA[0][lA0]);
        async_cp16(gA1, &sA[0][lA1]);
        #pragma unroll
        for (int j = 0; j < 8; ++j)
            async_cp16(gB0 + (size_t)(32 * j) * NN, &sB[0][lB0 + 32 * j * LDB]);
    }

    int buf = 0;
    for (int t = 0; t < NSTAGE; ++t) {
        if (t + 1 < NSTAGE) {
            const int ts = t + 1;
            const int s  = ts >> 6;                 // 64 stages per support
            const size_t kb = (size_t)(ts & 63) * KSTG;
            const int nb = buf ^ 1;
            async_cp16(gA0 + s * sStrA + kb, &sA[nb][lA0]);
            async_cp16(gA1 + s * sStrA + kb, &sA[nb][lA1]);
            #pragma unroll
            for (int j = 0; j < 8; ++j)
                async_cp16(gB0 + s * sStrB + (size_t)(32 * j) * NN + kb,
                           &sB[nb][lB0 + 32 * j * LDB]);
            WAIT_ASYNC(10);                         // stage t finished
        } else {
            WAIT_ASYNC(0);
        }
        __syncthreads();

        #pragma unroll
        for (int ks = 0; ks < 2; ++ks) {            // two 32-K sub-steps
            FragAB af[2];
            #pragma unroll
            for (int r = 0; r < 2; ++r) {
                const unsigned short* Ap =
                    &sA[buf][(mi * 32 + r * 16 + l15) * LDA + ks * 32 + lhi * 8];
                af[r].q[0] = *(const uint4*)(Ap);
                af[r].q[1] = *(const uint4*)(Ap + 16);
            }
            const unsigned short* Bp =
                &sB[buf][(nj * 64 + l15) * LDB + ks * 32 + lhi * 8];
            #pragma unroll
            for (int tt = 0; tt < 4; ++tt) {
                FragAB bf;
                bf.q[0] = *(const uint4*)(Bp + tt * 16 * LDB);
                bf.q[1] = *(const uint4*)(Bp + tt * 16 * LDB + 16);
                acc[0][tt].v = WMMA_BF16(af[0].v, bf.v, acc[0][tt].v);
                acc[1][tt].v = WMMA_BF16(af[1].v, bf.v, acc[1][tt].v);
            }
        }
        __syncthreads();
        buf ^= 1;
    }

    // C/D layout: lanes 0-15 -> rows i (VGPR i), lanes 16-31 -> rows i+8.
    #pragma unroll
    for (int r = 0; r < 2; ++r) {
        #pragma unroll
        for (int i = 0; i < 8; ++i) {
            int row = mi * 32 + r * 16 + i + lhi * 8;
            size_t base = ((size_t)b * DM + row) * NN + wbase0 + nj * 64 + l15;
            outw[base +  0] = acc[r][0].f[i];
            outw[base + 16] = acc[r][1].f[i];
            outw[base + 32] = acc[r][2].f[i];
            outw[base + 48] = acc[r][3].f[i];
        }
    }
}

// ---------------------------------------------------------------------------
// Kernel 4: per-node epilogue (lin + PReLU + read, tuple outputs).
// ---------------------------------------------------------------------------
__global__ __launch_bounds__(256) void k_epilogue(const float* __restrict__ outw,
                                                  const float* __restrict__ h,
                                                  const float* __restrict__ b_mlp,
                                                  const float* __restrict__ W_lin,
                                                  const float* __restrict__ b_lin,
                                                  const float* __restrict__ W_read,
                                                  const float* __restrict__ b_read,
                                                  const float* __restrict__ prelu_a,
                                                  float* __restrict__ out) {
    const int b = blockIdx.y;
    const int w = blockIdx.x * 256 + threadIdx.x;
    const float alpha = prelu_a[0];
    const size_t out3 = (size_t)BB * NN;

    float oc[DM], hc[DM];
    #pragma unroll
    for (int c = 0; c < DM; ++c)
        oc[c] = outw[((size_t)b * DM + c) * NN + w] + b_mlp[c];
    #pragma unroll
    for (int c = 0; c < DM; ++c)
        hc[c] = h[((size_t)b * DM + c) * NN + w];

    float racc = b_read[0];
    for (int o = 0; o < DM; ++o) {
        float acc = b_lin[o];
        const float* wl = W_lin + o * (2 * DM);
        #pragma unroll
        for (int c = 0; c < DM; ++c) acc += wl[c] * oc[c];
        #pragma unroll
        for (int c = 0; c < DM; ++c) acc += wl[DM + c] * hc[c];
        float p = (acc >= 0.f) ? acc : alpha * acc;
        out[out3 + ((size_t)b * (2 * DM) + o) * NN + w] = p;
        racc += W_read[o] * p;
    }
    #pragma unroll
    for (int c = 0; c < DM; ++c) {
        racc += W_read[DM + c] * hc[c];
        out[out3 + ((size_t)b * (2 * DM) + DM + c) * NN + w] = hc[c];
    }
    out[(size_t)b * NN + w] = racc;
}

// ---------------------------------------------------------------------------
extern "C" void kernel_launch(void* const* d_in, const int* in_sizes, int n_in,
                              void* d_out, int out_size, void* d_ws, size_t ws_size,
                              hipStream_t stream) {
    const float* x      = (const float*)d_in[0];
    const float* h      = (const float*)d_in[1];
    const float* adj    = (const float*)d_in[2];
    const float* W_mlp  = (const float*)d_in[3];
    const float* b_mlp  = (const float*)d_in[4];
    const float* W_lin  = (const float*)d_in[5];
    const float* b_lin  = (const float*)d_in[6];
    const float* W_read = (const float*)d_in[7];
    const float* b_read = (const float*)d_in[8];
    const float* prelua = (const float*)d_in[9];
    float* out = (float*)d_out;

    unsigned short* adjb = (unsigned short*)d_ws;                    // S*N*N
    unsigned short* Yb   = adjb + (size_t)SS * NN * NN;              // B*S*64*N
    float*          outw = (float*)(Yb + (size_t)BB * SS * DM * NN); // B*64*N

    k_cvt_adj<<<dim3((SS * NN * (size_t)NN) / (256 * 8)), dim3(256), 0, stream>>>(adj, adjb);
    k_prep_y<<<dim3(NN / 128, BB), dim3(256), 0, stream>>>(x, h, W_mlp, Yb);
    k_gemm<<<dim3(NN / BN, BB), dim3(256), 0, stream>>>(Yb, adjb, outw);
    k_epilogue<<<dim3(NN / 256, BB), dim3(256), 0, stream>>>(
        outw, h, b_mlp, W_lin, b_lin, W_read, b_read, prelua, out);
}
